// SimpleRNN_14181982011784
// MI455X (gfx1250) — matrix operand order repaired and checked by reading, not compiled, not run on hardware
//
#include <hip/hip_runtime.h>
#include <hip/hip_bf16.h>

typedef __attribute__((ext_vector_type(16))) _Float16 v16h;
typedef __attribute__((ext_vector_type(8)))  float    v8f;

#define T_LEN 4096
#define B_SZ  512
#define H_DIM 32

__device__ __forceinline__ float tanh_fast(float v) {
#if __has_builtin(__builtin_amdgcn_tanhf)
  return __builtin_amdgcn_tanhf(v);           // v_tanh_f32 (TRANS pipe)
#else
  float e = __builtin_amdgcn_exp2f(v * 2.88539008177792681472f);
  return 1.0f - 2.0f * __builtin_amdgcn_rcpf(e + 1.0f);
#endif
}

// swap value with lane^16 (wave32)
__device__ __forceinline__ float xor16(float v) {
#if __has_builtin(__builtin_amdgcn_permlanex16)
  int i = __float_as_int(v);
  // identity lane selects -> lane L gets lane L^16 ; VALU op, no LDS counter
  return __int_as_float(__builtin_amdgcn_permlanex16(i, i, 0x76543210, 0xfedcba98, false, false));
#else
  return __int_as_float(__builtin_amdgcn_ds_swizzle(__float_as_int(v), 0x401f));
#endif
}

__global__ __launch_bounds__(32, 1)
void rnn_scan_kernel(const float* __restrict__ x, const float* __restrict__ h0,
                     const float* __restrict__ W_ih, const float* __restrict__ b_ih,
                     const float* __restrict__ W_hh, const float* __restrict__ b_hh,
                     const float* __restrict__ W_out, const float* __restrict__ b_out,
                     float* __restrict__ outp, float* __restrict__ hstate) {
  const int lane = threadIdx.x & 31;
  const int bl   = lane & 15;   // batch row within tile (WMMA N / A-layout M index)
  const int hi   = lane >> 4;   // lane half (selects K-chunk / D row-offset)
  const int b    = blockIdx.x * 16 + bl;

  // Unit permutation so that D(C-layout) == next hB(B-layout) with no lane swap:
  //   A0 rows: r<8 -> unit r   ; r>=8 -> unit r+8   (units {0..7,16..23})
  //   A1 rows: r<8 -> unit r+8 ; r>=8 -> unit r+16  (units {8..15,24..31})
  const int rowA0 = (bl < 8) ? bl : bl + 8;
  const int rowA1 = (bl < 8) ? bl + 8 : bl + 16;
  const int kb    = hi * 16;    // contiguous k order (same as hB): lanes<16 hold K 0..15, lanes>=16 hold 16..31
  v16h a0, a1;
  #pragma unroll
  for (int j = 0; j < 8; ++j) {
    a0[j]     = (_Float16)W_hh[rowA0 * H_DIM + kb + j];
    a0[j + 8] = (_Float16)W_hh[rowA0 * H_DIM + kb + 8 + j];
    a1[j]     = (_Float16)W_hh[rowA1 * H_DIM + kb + j];
    a1[j + 8] = (_Float16)W_hh[rowA1 * H_DIM + kb + 8 + j];
  }

  // Per-lane unit sets delivered by D0/D1: U0(v)=v+16*hi, U1(v)=8+v+16*hi
  float wih0[8], wih1[8], bia0[8], bia1[8], wo0[8], wo1[8];
  #pragma unroll
  for (int v = 0; v < 8; ++v) {
    const int i0 = v + 16 * hi;
    const int i1 = 8 + v + 16 * hi;
    wih0[v] = W_ih[i0];            wih1[v] = W_ih[i1];
    bia0[v] = b_ih[i0] + b_hh[i0]; bia1[v] = b_ih[i1] + b_hh[i1];
    wo0[v]  = W_out[i0];           wo1[v]  = W_out[i1];
  }
  const float bo = b_out[0];

  // Initial hidden state as B fragment: lane half hi holds k = 16*hi + j for batch bl
  v16h hB;
  #pragma unroll
  for (int j = 0; j < 16; ++j)
    hB[j] = (_Float16)h0[b * H_DIM + 16 * hi + j];

  const float* xrow = x + (size_t)b * T_LEN;
  float*       orow = outp + (size_t)b * T_LEN;

  float t0[8], t1[8];
  // software-pipelined x stream: consume xc (loaded last block), load next block now
  float4 xc = *(const float4*)(xrow);
  for (int tb = 0; tb < T_LEN; tb += 4) {
    const int tn = (tb + 4 < T_LEN) ? (tb + 4) : tb;
    const float4 xn = *(const float4*)(xrow + tn);       // waited on only next block
    __builtin_prefetch(xrow + tb + 64, 0, 0);            // stage lines ~2 ahead into cache
    const float xs[4] = {xc.x, xc.y, xc.z, xc.w};
    #pragma unroll
    for (int u = 0; u < 4; ++u) {
      const float xv = xs[u];
      // input projection + biases as WMMA C operand (off the serial chain)
      v8f c0, c1;
      #pragma unroll
      for (int v = 0; v < 8; ++v) {
        c0[v] = fmaf(xv, wih0[v], bia0[v]);
        c1[v] = fmaf(xv, wih1[v], bia1[v]);
      }
      v8f d0 = __builtin_amdgcn_wmma_f32_16x16x32_f16(false, a0, false, hB,
                                                      (short)0, c0, false, false);
      v8f d1 = __builtin_amdgcn_wmma_f32_16x16x32_f16(false, a1, false, hB,
                                                      (short)0, c1, false, false);
      #pragma unroll
      for (int v = 0; v < 8; ++v) {
        t0[v] = tanh_fast(d0[v]);
        t1[v] = tanh_fast(d1[v]);
      }
      // next-step B fragment: position 16*hi+j ; j<8 from D0, j>=8 from D1 (no lane swap)
      #pragma unroll
      for (int v = 0; v < 8; ++v) {
        hB[v]     = (_Float16)t0[v];
        hB[v + 8] = (_Float16)t1[v];
      }
      // output projection (scalar per (b,t)) — off the serial chain.
      // po + xor16(po) is bitwise identical in lanes L and L^16, so both halves
      // store the same value to the same address: no exec-mask branch needed.
      float po = 0.0f;
      #pragma unroll
      for (int v = 0; v < 8; ++v)
        po = fmaf(t0[v], wo0[v], fmaf(t1[v], wo1[v], po));
      po += xor16(po);
      orow[tb + u] = po + bo;
    }
    xc = xn;
  }

  // final hidden state: lane holds 16 contiguous units starting at 16*hi
  #pragma unroll
  for (int v = 0; v < 8; ++v) {
    hstate[b * H_DIM + 16 * hi + v]     = t0[v];
    hstate[b * H_DIM + 16 * hi + 8 + v] = t1[v];
  }
}

extern "C" void kernel_launch(void* const* d_in, const int* in_sizes, int n_in,
                              void* d_out, int out_size, void* d_ws, size_t ws_size,
                              hipStream_t stream) {
  (void)in_sizes; (void)n_in; (void)out_size; (void)d_ws; (void)ws_size;
  const float* x     = (const float*)d_in[0];
  const float* h0    = (const float*)d_in[1];
  const float* W_ih  = (const float*)d_in[2];
  const float* b_ih  = (const float*)d_in[3];
  const float* W_hh  = (const float*)d_in[4];
  const float* b_hh  = (const float*)d_in[5];
  const float* W_out = (const float*)d_in[6];
  const float* b_out = (const float*)d_in[7];
  float* outp   = (float*)d_out;                      // [B,T,O] flat
  float* hstate = outp + (size_t)B_SZ * T_LEN;        // [1,B,H] flat

  rnn_scan_kernel<<<B_SZ / 16, 32, 0, stream>>>(x, h0, W_ih, b_ih, W_hh, b_hh,
                                                W_out, b_out, outp, hstate);
}
